// GNNReID_4277787427363
// MI455X (gfx1250) — matrix operand order repaired and checked by reading, not compiled, not run on hardware
//
#include <hip/hip_runtime.h>
#include <hip/hip_bf16.h>

#define N_NODES 50000
#define N_EDGES 800000
#define DIM     128
#define NHEAD   8
#define HDIM    16
#define FFN     512
#define LN_EPS  1e-5f

typedef __attribute__((ext_vector_type(16))) _Float16 v16h;
typedef __attribute__((ext_vector_type(8)))  _Float16 v8h;
typedef __attribute__((ext_vector_type(4)))  _Float16 v4h;
typedef __attribute__((ext_vector_type(8)))  float    v8f;
typedef __attribute__((ext_vector_type(4)))  int      v4i;

#if defined(__has_builtin)
#if __has_builtin(__builtin_amdgcn_global_load_async_to_lds_b128)
#define HAVE_ASYNC_LDS 1
#endif
#endif

__device__ __forceinline__ void wait_async_zero() {
#if defined(__has_builtin) && __has_builtin(__builtin_amdgcn_s_wait_asynccnt)
    __builtin_amdgcn_s_wait_asynccnt(0);
#else
    asm volatile("s_wait_asynccnt 0x0" ::: "memory");
#endif
}

// ---------------- helpers: float <-> order-preserving uint (for atomicMax) ----
__device__ __forceinline__ unsigned f2ord(float f) {
    unsigned u = __float_as_uint(f);
    return (u & 0x80000000u) ? ~u : (u | 0x80000000u);
}
__device__ __forceinline__ float ord2f(unsigned u) {
    u = (u & 0x80000000u) ? (u & 0x7FFFFFFFu) : ~u;
    return __uint_as_float(u);
}

// ---------------- f32 -> f16 convert (weights, agg) --------------------------
__global__ void cvt_f32_f16_kernel(const float* __restrict__ s,
                                   _Float16* __restrict__ d, int n) {
    int i = blockIdx.x * blockDim.x + threadIdx.x;
    if (i < n) d[i] = (_Float16)s[i];
}

// ---------------- LayerNorm (ddof=1) -> f16, one wave32 per 128-wide row -----
__global__ void ln_to_f16_kernel(const float* __restrict__ x,
                                 const float* __restrict__ w,
                                 const float* __restrict__ b,
                                 _Float16* __restrict__ y, int nrows) {
    int wid  = threadIdx.x >> 5;
    int lane = threadIdx.x & 31;
    int row  = blockIdx.x * (blockDim.x >> 5) + wid;
    if (row >= nrows) return;
    float4 xv = ((const float4*)(x + (size_t)row * DIM))[lane];
    float s  = xv.x + xv.y + xv.z + xv.w;
    float ss = xv.x*xv.x + xv.y*xv.y + xv.z*xv.z + xv.w*xv.w;
    #pragma unroll
    for (int m = 16; m >= 1; m >>= 1) {
        s  += __shfl_xor(s,  m, 32);
        ss += __shfl_xor(ss, m, 32);
    }
    float mu  = s * (1.0f / DIM);
    float var = (ss - (float)DIM * mu * mu) * (1.0f / (DIM - 1));
    float sd  = sqrtf(fmaxf(var, 0.0f));
    float inv = 1.0f / (sd + LN_EPS);
    float4 wv = ((const float4*)w)[lane];
    float4 bv = ((const float4*)b)[lane];
    v4h o;
    o[0] = (_Float16)((xv.x - mu) * inv * wv.x + bv.x);
    o[1] = (_Float16)((xv.y - mu) * inv * wv.y + bv.y);
    o[2] = (_Float16)((xv.z - mu) * inv * wv.z + bv.z);
    o[3] = (_Float16)((xv.w - mu) * inv * wv.w + bv.w);
    *(v4h*)(y + (size_t)row * DIM + lane * 4) = o;
}

// ---------------- WMMA GEMM: Y[nrows,M] = A[nrows,K] @ W[M,K]^T + bias -------
// Grid: (ceil(rowTiles/4), M/64). Block = 128 threads = 4 waves.
// The block's shared 64-row weight tile is async-copied to LDS once
// (global_load_async_to_lds_b128 + s_wait_asynccnt + barrier), then each wave
// computes a 16x64 output tile with 4 f32 16x16 accumulators, B from LDS.
__global__ void wmma_gemm_kernel(const _Float16* __restrict__ A,
                                 const _Float16* __restrict__ W,
                                 const float* __restrict__ bias,
                                 const float* __restrict__ resid,
                                 float* __restrict__ outF,
                                 _Float16* __restrict__ outH,
                                 int nrows, int K, int M, int relu) {
    extern __shared__ _Float16 ldsW[];   // 64 x K f16, row-major
    const int wid  = threadIdx.x >> 5;
    const int lane = threadIdx.x & 31;
    const int half = lane >> 4;          // lanes 16-31 handle upper K/M halves
    const int l16  = lane & 15;
    const int rowTiles = nrows >> 4;
    const int colBase  = blockIdx.y * 64;

    // ---- cooperative async copy of W rows [colBase, colBase+64) into LDS ----
    const _Float16* gsrc = W + (size_t)colBase * K;   // contiguous 64*K f16
    const int chunks = (64 * K) >> 3;                 // 16-byte chunks
    for (int i = threadIdx.x; i < chunks; i += blockDim.x) {
#if defined(HAVE_ASYNC_LDS)
        __builtin_amdgcn_global_load_async_to_lds_b128(
            (v4i*)(gsrc + (size_t)i * 8),
            (v4i*)(ldsW + (size_t)i * 8), 0, 0);
#else
        *(v8h*)(ldsW + (size_t)i * 8) = *(const v8h*)(gsrc + (size_t)i * 8);
#endif
    }
#if defined(HAVE_ASYNC_LDS)
    wait_async_zero();
#endif
    __syncthreads();

    const int rt = blockIdx.x * 4 + wid;   // wave-uniform guard: EXEC all-1s
    if (rt < rowTiles) {
        const int arow = rt * 16 + l16;
        v8f acc[4] = {};
        const _Float16* aBase = A + (size_t)arow * K + half * 8;

        for (int k0 = 0; k0 < K; k0 += 32) {
            __builtin_prefetch(aBase + k0 + 64, 0, 3);
            // A frag (16-bit A 16x32): lanes<16 K{0..7,16..23}, lanes>=16 K{8..15,24..31}
            v8h a0 = *(const v8h*)(aBase + k0);
            v8h a1 = *(const v8h*)(aBase + k0 + 16);
            v16h af;
            #pragma unroll
            for (int i = 0; i < 8; ++i) { af[i] = a0[i]; af[i + 8] = a1[i]; }
            #pragma unroll
            for (int c = 0; c < 4; ++c) {
                // B frag from LDS: column (= W row) c*16+l16, lanes<16 K{0..7,16..23}, lanes>=16 K{8..15,24..31}
                const _Float16* bp = ldsW + (size_t)(c * 16 + l16) * K + k0 + half * 8;
                v8h b0 = *(const v8h*)bp;
                v8h b1 = *(const v8h*)(bp + 16);
                v16h bf;
                #pragma unroll
                for (int i = 0; i < 8; ++i) { bf[i] = b0[i]; bf[i + 8] = b1[i]; }
                acc[c] = __builtin_amdgcn_wmma_f32_16x16x32_f16(
                    false, af, false, bf, (short)0, acc[c], false, false);
            }
        }

        #pragma unroll
        for (int c = 0; c < 4; ++c) {
            int col = colBase + c * 16 + l16;
            float bb = bias ? bias[col] : 0.0f;
            #pragma unroll
            for (int j = 0; j < 8; ++j) {
                int r = rt * 16 + half * 8 + j;  // C/D layout: VGPR j -> M = j + 8*half
                float v = acc[c][j] + bb;
                if (resid) v += resid[(size_t)r * M + col];
                if (relu)  v = fmaxf(v, 0.0f);
                if (outF)  outF[(size_t)r * M + col] = v;
                if (outH)  outH[(size_t)r * M + col] = (_Float16)v;
            }
        }
    }
}

// ---------------- Edge attention: scores + segment max -----------------------
__global__ void edge_scores_kernel(const int* __restrict__ ei,
                                   const float* __restrict__ q,
                                   const float* __restrict__ k,
                                   float* __restrict__ scores,
                                   unsigned* __restrict__ smax) {
    int t = blockIdx.x * blockDim.x + threadIdx.x;
    if (t >= N_EDGES * NHEAD) return;
    int e = t >> 3, h = t & 7;
    int r = ei[2 * e], c = ei[2 * e + 1];
    const float* qp = q + (size_t)c * DIM + h * HDIM;
    const float* kp = k + (size_t)r * DIM + h * HDIM;
    float s = 0.0f;
    #pragma unroll
    for (int d = 0; d < HDIM; ++d) s += qp[d] * kp[d];
    s *= 0.25f;  // 1/sqrt(16)
    scores[t] = s;
    atomicMax(&smax[r * NHEAD + h], f2ord(s));
}

// ---------------- exp(score - max), segment sum of denominators --------------
__global__ void edge_exp_kernel(const int* __restrict__ ei,
                                float* __restrict__ scores,
                                const unsigned* __restrict__ smax,
                                float* __restrict__ denom) {
    int t = blockIdx.x * blockDim.x + threadIdx.x;
    if (t >= N_EDGES * NHEAD) return;
    int e = t >> 3, h = t & 7;
    int r = ei[2 * e];
    float m  = ord2f(smax[r * NHEAD + h]);
    float ex = __expf(scores[t] - m);
    scores[t] = ex;
    atomicAdd(&denom[r * NHEAD + h], ex);
}

// ---------------- alpha * v[row] scatter-added into agg[col] -----------------
__global__ void edge_msg_kernel(const int* __restrict__ ei,
                                const float* __restrict__ scores,
                                const float* __restrict__ denom,
                                const float* __restrict__ v,
                                float* __restrict__ agg) {
    int t = blockIdx.x * blockDim.x + threadIdx.x;
    if (t >= N_EDGES * NHEAD) return;
    int e = t >> 3, h = t & 7;
    int r = ei[2 * e], c = ei[2 * e + 1];
    float alpha = scores[t] / denom[r * NHEAD + h];
    const float* vp = v + (size_t)r * DIM + h * HDIM;
    float* ap = agg + (size_t)c * DIM + h * HDIM;
    #pragma unroll
    for (int d = 0; d < HDIM; ++d) atomicAdd(&ap[d], alpha * vp[d]);
}

// =============================================================================
extern "C" void kernel_launch(void* const* d_in, const int* in_sizes, int n_in,
                              void* d_out, int out_size, void* d_ws, size_t ws_size,
                              hipStream_t stream) {
    const float* feats = (const float*)d_in[0];
    const int*   ei    = (const int*)d_in[1];
    const float* Wq    = (const float*)d_in[2];
    const float* bq    = (const float*)d_in[3];
    const float* Wk    = (const float*)d_in[4];
    const float* bk    = (const float*)d_in[5];
    const float* Wv    = (const float*)d_in[6];
    const float* bv    = (const float*)d_in[7];
    const float* Wo    = (const float*)d_in[8];
    const float* bo    = (const float*)d_in[9];
    const float* ln1w  = (const float*)d_in[10];
    const float* ln1b  = (const float*)d_in[11];
    const float* ln2w  = (const float*)d_in[12];
    const float* ln2b  = (const float*)d_in[13];
    const float* W1    = (const float*)d_in[14];
    const float* b1    = (const float*)d_in[15];
    const float* W2    = (const float*)d_in[16];
    const float* b2    = (const float*)d_in[17];
    float* out = (float*)d_out;

    char*  ws  = (char*)d_ws;
    size_t off = 0;
    auto alloc = [&](size_t bytes) -> char* {
        char* p = ws + off;
        off = (off + bytes + 255) & ~(size_t)255;
        return p;
    };

    const size_t ND = (size_t)N_NODES * DIM;
    const size_t NF = (size_t)N_NODES * FFN;
    const size_t EH = (size_t)N_EDGES * NHEAD;

    _Float16* x2h   = (_Float16*)alloc(ND * 2);
    _Float16* wqh   = (_Float16*)alloc((size_t)DIM * DIM * 2);
    _Float16* wkh   = (_Float16*)alloc((size_t)DIM * DIM * 2);
    _Float16* wvh   = (_Float16*)alloc((size_t)DIM * DIM * 2);
    _Float16* woh   = (_Float16*)alloc((size_t)DIM * DIM * 2);
    _Float16* w1h   = (_Float16*)alloc((size_t)FFN * DIM * 2);
    _Float16* w2h   = (_Float16*)alloc((size_t)DIM * FFN * 2);
    float*    q     = (float*)alloc(ND * 4);
    float*    k     = (float*)alloc(ND * 4);
    float*    v     = (float*)alloc(ND * 4);
    float*    scr   = (float*)alloc(EH * 4);
    unsigned* smax  = (unsigned*)alloc((size_t)N_NODES * NHEAD * 4);
    float*    denom = (float*)alloc((size_t)N_NODES * NHEAD * 4);
    float*    agg   = (float*)alloc(ND * 4);
    _Float16* aggh  = (_Float16*)alloc(ND * 2);
    float*    f2    = (float*)alloc(ND * 4);     // feats + attention
    _Float16* x2bh  = (_Float16*)alloc(ND * 2);
    _Float16* h1h   = (_Float16*)alloc(NF * 2);

    (void)hipMemsetAsync(smax,  0, (size_t)N_NODES * NHEAD * 4, stream);
    (void)hipMemsetAsync(denom, 0, (size_t)N_NODES * NHEAD * 4, stream);
    (void)hipMemsetAsync(agg,   0, ND * 4, stream);

    // weights -> f16
    const int dd = DIM * DIM, fd = FFN * DIM;
    cvt_f32_f16_kernel<<<(dd + 255) / 256, 256, 0, stream>>>(Wq, wqh, dd);
    cvt_f32_f16_kernel<<<(dd + 255) / 256, 256, 0, stream>>>(Wk, wkh, dd);
    cvt_f32_f16_kernel<<<(dd + 255) / 256, 256, 0, stream>>>(Wv, wvh, dd);
    cvt_f32_f16_kernel<<<(dd + 255) / 256, 256, 0, stream>>>(Wo, woh, dd);
    cvt_f32_f16_kernel<<<(fd + 255) / 256, 256, 0, stream>>>(W1, w1h, fd);
    cvt_f32_f16_kernel<<<(fd + 255) / 256, 256, 0, stream>>>(W2, w2h, fd);

    // LN1 -> x2 (f16)
    ln_to_f16_kernel<<<N_NODES / 8, 256, 0, stream>>>(feats, ln1w, ln1b, x2h, N_NODES);

    // GEMM grids: x = ceil(rowTiles/4) blocks of 4 waves, y = output-col tiles
    const int rowTiles  = N_NODES / 16;               // 3125
    const int rowBlocks = (rowTiles + 3) / 4;         // 782
    const size_t lds128 = (size_t)64 * 128 * 2;       // 16 KB
    const size_t lds512 = (size_t)64 * 512 * 2;       // 64 KB
    dim3 gD(rowBlocks, DIM / 64);                     // (782, 2)
    dim3 gF1(rowBlocks, FFN / 64);                    // (782, 8)

    wmma_gemm_kernel<<<gD, 128, lds128, stream>>>(x2h, wqh, bq, nullptr, q, nullptr, N_NODES, DIM, DIM, 0);
    wmma_gemm_kernel<<<gD, 128, lds128, stream>>>(x2h, wkh, bk, nullptr, k, nullptr, N_NODES, DIM, DIM, 0);
    wmma_gemm_kernel<<<gD, 128, lds128, stream>>>(x2h, wvh, bv, nullptr, v, nullptr, N_NODES, DIM, DIM, 0);

    // edge attention
    int blksE = (int)((EH + 255) / 256);
    edge_scores_kernel<<<blksE, 256, 0, stream>>>(ei, q, k, scr, smax);
    edge_exp_kernel   <<<blksE, 256, 0, stream>>>(ei, scr, smax, denom);
    edge_msg_kernel   <<<blksE, 256, 0, stream>>>(ei, scr, denom, v, agg);

    // agg -> f16, then output projection with residual: f2 = feats + agg@Wo^T + bo
    cvt_f32_f16_kernel<<<(int)((ND + 255) / 256), 256, 0, stream>>>(agg, aggh, (int)ND);
    wmma_gemm_kernel<<<gD, 128, lds128, stream>>>(aggh, woh, bo, feats, f2, nullptr, N_NODES, DIM, DIM, 0);

    // LN2 -> x2b (f16)
    ln_to_f16_kernel<<<N_NODES / 8, 256, 0, stream>>>(f2, ln2w, ln2b, x2bh, N_NODES);

    // FFN1: h1 = relu(x2b@W1^T + b1), stored f16 (feeds FFN2 directly)
    wmma_gemm_kernel<<<gF1, 128, lds128, stream>>>(x2bh, w1h, b1, nullptr, nullptr, h1h, N_NODES, DIM, FFN, 1);

    // FFN2: out = relu(f2 + h1@W2^T + b2)
    wmma_gemm_kernel<<<gD, 128, lds512, stream>>>(h1h, w2h, b2, f2, out, nullptr, N_NODES, FFN, DIM, 1);
}
